// MultiHeadAttention_80066780332377
// MI455X (gfx1250) — compile-verified, matrix operand order checked by reading, not run on hardware
//
#include <hip/hip_runtime.h>
#include <hip/hip_bf16.h>
#include <stdint.h>

// ---------- constants ----------
#define DM    1024
#define NHEAD 16
#define HD    64
#define SLEN  2048
#define BATCH 2
#define MROWS (BATCH * SLEN)   // 4096

typedef __attribute__((ext_vector_type(16))) __bf16 bf16x16;
typedef __attribute__((ext_vector_type(8)))  __bf16 bf16x8;
typedef __attribute__((ext_vector_type(8)))  float  v8f;

__device__ __forceinline__ __bf16 f2bf(float f) {
    unsigned int u = __float_as_uint(f);
    unsigned int r = (u + 0x7FFFu + ((u >> 16) & 1u)) >> 16;
    unsigned short s = (unsigned short)r;
    return __builtin_bit_cast(__bf16, s);
}

// ---------- fp32 -> bf16 convert ----------
__global__ void cvt_bf16_kernel(const float* __restrict__ src,
                                __bf16* __restrict__ dst, int n) {
    int tid = blockIdx.x * blockDim.x + threadIdx.x;
    if (tid < n) dst[tid] = f2bf(src[tid]);
}

// ---------- WMMA GEMM: C[M,N] = A[M,K] * W[N,K]^T + bias ----------
// block = 256 threads (8 waves); block tile 64(M) x 128(N); wave tile 16 x 64.
// A-tile (64x32 bf16) is staged into LDS with double-buffered
// global_load_async_to_lds_b128 (ASYNCcnt pipeline); B goes straight to VGPRs.
__global__ __launch_bounds__(256) void gemm_bf16_kernel(
        const __bf16* __restrict__ A, const __bf16* __restrict__ W,
        const float* __restrict__ bias, float* __restrict__ C,
        int M, int N, int Kd) {
    __shared__ __bf16 at[2][64][32];

    const int tid  = threadIdx.x;
    const int wave = tid >> 5;
    const int lane = tid & 31;
    const int wm = wave & 3;          // 4 M-subtiles
    const int wn = wave >> 2;         // 2 N-subtiles
    const int m_blk = blockIdx.x * 64;
    const int m0 = m_blk + wm * 16;
    const int n0 = blockIdx.y * 128 + wn * 64;
    const int n  = lane & 15;
    const int kc = (lane >> 4) * 8;   // K-offset within 32-chunk for this half-wave

    // async-copy work assignment: thread tid moves one 16B chunk per k-step
    const int crow = tid >> 2;              // 0..63 (tile row)
    const int ccol = (tid & 3) * 8;         // 0,8,16,24 (tile col)
    const unsigned voff = (unsigned)(((m_blk + crow) * Kd + ccol) * 2);
    unsigned ldsaddr[2];
    ldsaddr[0] = (unsigned)(uintptr_t)&at[0][crow][ccol];
    ldsaddr[1] = (unsigned)(uintptr_t)&at[1][crow][ccol];

    v8f acc[4];
#pragma unroll
    for (int t = 0; t < 4; ++t) acc[t] = v8f{0.f,0.f,0.f,0.f,0.f,0.f,0.f,0.f};

    const __bf16* brow[4];
#pragma unroll
    for (int t = 0; t < 4; ++t) brow[t] = W + (size_t)(n0 + t * 16 + n) * Kd;

    const int nkb = Kd / 32;
    const unsigned long long sbase = (unsigned long long)(uintptr_t)A;

    // prime the pipeline: tile 0 -> buffer 0
    asm volatile("global_load_async_to_lds_b128 %0, %1, %2"
                 :: "v"(ldsaddr[0]), "v"(voff), "s"(sbase) : "memory");

    for (int kb = 0; kb < nkb; ++kb) {
        // current tile's async copies done (per wave), then workgroup-visible
        asm volatile("s_wait_asynccnt 0x0" ::: "memory");
        __syncthreads();

        // kick off next tile into the other buffer (overlaps with compute)
        if (kb + 1 < nkb) {
            const unsigned long long nb = sbase + 64ull * (unsigned long long)(kb + 1);
            asm volatile("global_load_async_to_lds_b128 %0, %1, %2"
                         :: "v"(ldsaddr[(kb + 1) & 1]), "v"(voff), "s"(nb) : "memory");
        }

        const int k = kb * 32;

        // A fragment from LDS
        bf16x16 af;
        {
            const __bf16* prow = &at[kb & 1][wm * 16 + (lane & 15)][0];
            bf16x8 lo = *(const bf16x8*)(prow + kc);
            bf16x8 hi = *(const bf16x8*)(prow + kc + 16);
#pragma unroll
            for (int u = 0; u < 8; ++u) { af[u] = lo[u]; af[8 + u] = hi[u]; }
        }

        // preload all 4 B fragments, then burst the WMMAs
        bf16x16 bfm[4];
#pragma unroll
        for (int t = 0; t < 4; ++t) {
            bf16x8 blo = *(const bf16x8*)(brow[t] + k + kc);
            bf16x8 bhi = *(const bf16x8*)(brow[t] + k + kc + 16);
#pragma unroll
            for (int u = 0; u < 8; ++u) { bfm[t][u] = blo[u]; bfm[t][8 + u] = bhi[u]; }
        }
        __builtin_prefetch(brow[0] + k + 128, 0, 1);
#pragma unroll
        for (int t = 0; t < 4; ++t)
            acc[t] = __builtin_amdgcn_wmma_f32_16x16x32_bf16(
                false, af, false, bfm[t], (short)0, acc[t], false, false);
    }

    const int mb = (lane >> 4) * 8;
#pragma unroll
    for (int t = 0; t < 4; ++t) {
        const int col = n0 + t * 16 + n;
        const float bv = bias[col];
#pragma unroll
        for (int r = 0; r < 8; ++r)
            C[(size_t)(m0 + mb + r) * N + col] = acc[t][r] + bv;
    }
}

// ---------- RoPE + reshape [B,S,D] fp32 -> [B,H,S,Dh] bf16 ----------
__global__ void rope_reshape_kernel(const float* __restrict__ lin,
                                    __bf16* __restrict__ dst) {
    int tid = blockIdx.x * blockDim.x + threadIdx.x; // B*H*S*(HD/2) = 2^21
    int i  = tid & 31;            // pair index 0..31
    int s  = (tid >> 5) & (SLEN - 1);
    int bh = tid >> 16;           // 0..31
    int b = bh >> 4, h = bh & 15;
    const float* src = lin + ((size_t)(b * SLEN + s)) * DM + h * HD + 2 * i;
    float xr = src[0], xi = src[1];
    // inv_freq = ROPE_BASE^(-2i/Dh); log(10000)=9.2103403719761836
    float inv = __expf(-9.2103403719761836f * (float)(2 * i) * (1.0f / 64.0f));
    float ang = (float)s * inv;
    float sn, cs;
    __sincosf(ang, &sn, &cs);
    __bf16* d = dst + ((size_t)(b * NHEAD + h) * SLEN + s) * HD + 2 * i;
    d[0] = f2bf(xr * cs - xi * sn);
    d[1] = f2bf(xr * sn + xi * cs);
}

// ---------- V reshape+transpose [B,S,D] fp32 -> [B,H,Dh,S] bf16 ----------
__global__ void v_transpose_kernel(const float* __restrict__ lin,
                                   __bf16* __restrict__ Vt) {
    int tid = blockIdx.x * blockDim.x + threadIdx.x; // B*S*D = 2^22
    int d = tid & (DM - 1);
    int s = (tid >> 10) & (SLEN - 1);
    int b = tid >> 21;
    int h = d >> 6, dh = d & 63;
    Vt[(((size_t)(b * NHEAD + h) * HD) + dh) * SLEN + s] = f2bf(lin[tid]);
}

// ---------- Flash attention ----------
// grid (B*H, S/16/4), block 128 = 4 waves; each wave: one 16-row q tile.
// Q,K: [B,H,S,Dh] bf16; Vt: [B,H,Dh,S] bf16; O: [B,S,H*Dh] bf16.
__global__ __launch_bounds__(128) void attn_kernel(
        const __bf16* __restrict__ Q, const __bf16* __restrict__ K,
        const __bf16* __restrict__ Vt, __bf16* __restrict__ O) {
    __shared__ __bf16 ldsP[4][16][32];

    const int wave = threadIdx.x >> 5;
    const int lane = threadIdx.x & 31;
    const int bh = blockIdx.x;
    const int b = bh >> 4, h = bh & 15;
    const int qt = blockIdx.y * 4 + wave;
    const int q0 = qt * 16;

    const __bf16* Qb = Q + ((size_t)(b * NHEAD + h) * SLEN + q0) * HD;
    const __bf16* Kb = K + (size_t)(b * NHEAD + h) * SLEN * HD;
    const __bf16* Vb = Vt + (size_t)(b * NHEAD + h) * HD * SLEN;

    const int n  = lane & 15;
    const int g  = lane >> 4;
    const int kc = g * 8;

    // Q A-fragments for the two K-chunks (d = 0..31, 32..63)
    bf16x16 qf[2];
    {
        const __bf16* qrow = Qb + (size_t)(lane & 15) * HD;
#pragma unroll
        for (int j = 0; j < 2; ++j) {
            bf16x8 lo = *(const bf16x8*)(qrow + j * 32 + kc);
            bf16x8 hi = *(const bf16x8*)(qrow + j * 32 + kc + 16);
#pragma unroll
            for (int u = 0; u < 8; ++u) { qf[j][u] = lo[u]; qf[j][8 + u] = hi[u]; }
        }
    }

    float m_s[8], l_s[8];
    v8f acc[4];
#pragma unroll
    for (int r = 0; r < 8; ++r) { m_s[r] = -1e30f; l_s[r] = 0.f; }
#pragma unroll
    for (int t = 0; t < 4; ++t) acc[t] = v8f{0.f,0.f,0.f,0.f,0.f,0.f,0.f,0.f};

    const float scale = 0.125f; // 1/sqrt(64)
    const int nkb = (q0 + 16 + 31) / 32;

    for (int kb = 0; kb < nkb; ++kb) {
        const int kbase = kb * 32;

        // ---- scores: two 16x16 tiles, K = Dh = 64 (2 wmma each) ----
        v8f sc[2];
#pragma unroll
        for (int c = 0; c < 2; ++c) {
            v8f z = v8f{0.f,0.f,0.f,0.f,0.f,0.f,0.f,0.f};
            const __bf16* krow = Kb + (size_t)(kbase + c * 16 + n) * HD;
#pragma unroll
            for (int j = 0; j < 2; ++j) {
                bf16x8 lo = *(const bf16x8*)(krow + j * 32 + kc);
                bf16x8 hi = *(const bf16x8*)(krow + j * 32 + kc + 16);
                bf16x16 bfm;
#pragma unroll
                for (int u = 0; u < 8; ++u) { bfm[u] = lo[u]; bfm[8 + u] = hi[u]; }
                z = __builtin_amdgcn_wmma_f32_16x16x32_bf16(
                    false, qf[j], false, bfm, (short)0, z, false, false);
            }
            sc[c] = z;
        }

        // ---- online softmax (rows live across 16-lane halves) ----
#pragma unroll
        for (int r = 0; r < 8; ++r) {
            const int rowq = q0 + g * 8 + r;
            float s0 = sc[0][r] * scale;
            float s1 = sc[1][r] * scale;
            if (kbase + n > rowq)      s0 = -1e30f;
            if (kbase + 16 + n > rowq) s1 = -1e30f;
            float mx = fmaxf(s0, s1);
#pragma unroll
            for (int msk = 8; msk >= 1; msk >>= 1)
                mx = fmaxf(mx, __shfl_xor(mx, msk, 32));
            const float mnew  = fmaxf(m_s[r], mx);
            const float alpha = __expf(m_s[r] - mnew);
            const float p0 = __expf(s0 - mnew);
            const float p1 = __expf(s1 - mnew);
            float rs = p0 + p1;
#pragma unroll
            for (int msk = 8; msk >= 1; msk >>= 1)
                rs += __shfl_xor(rs, msk, 32);
            l_s[r] = l_s[r] * alpha + rs;
            m_s[r] = mnew;
#pragma unroll
            for (int t = 0; t < 4; ++t) acc[t][r] *= alpha;
            ldsP[wave][g * 8 + r][n]      = f2bf(p0);
            ldsP[wave][g * 8 + r][16 + n] = f2bf(p1);
        }

        __builtin_amdgcn_wave_barrier();
        asm volatile("s_wait_dscnt 0x0" ::: "memory");

        // ---- re-shape P (C-layout) into an A-fragment via LDS ----
        bf16x16 pf;
        {
            const __bf16* prow = &ldsP[wave][lane & 15][0];
            bf16x8 lo = *(const bf16x8*)(prow + kc);
            bf16x8 hi = *(const bf16x8*)(prow + kc + 16);
#pragma unroll
            for (int u = 0; u < 8; ++u) { pf[u] = lo[u]; pf[8 + u] = hi[u]; }
        }
        __builtin_amdgcn_wave_barrier();

        // ---- out += P (16x32) x V (32x64) ----
#pragma unroll
        for (int t = 0; t < 4; ++t) {
            const __bf16* vcol = Vb + (size_t)(t * 16 + n) * SLEN + kbase;
            bf16x8 lo = *(const bf16x8*)(vcol + kc);
            bf16x8 hi = *(const bf16x8*)(vcol + kc + 16);
            bf16x16 bfm;
#pragma unroll
            for (int u = 0; u < 8; ++u) { bfm[u] = lo[u]; bfm[8 + u] = hi[u]; }
            acc[t] = __builtin_amdgcn_wmma_f32_16x16x32_bf16(
                false, pf, false, bfm, (short)0, acc[t], false, false);
        }
    }

    // ---- normalize + store [B,S,H*Dh] bf16 ----
#pragma unroll
    for (int r = 0; r < 8; ++r) {
        const float inv = 1.0f / l_s[r];
        const int rowq = q0 + g * 8 + r;
        __bf16* orow = O + ((size_t)(b * SLEN) + rowq) * DM + h * HD;
#pragma unroll
        for (int t = 0; t < 4; ++t)
            orow[t * 16 + n] = f2bf(acc[t][r] * inv);
    }
}

// ---------- host ----------
extern "C" void kernel_launch(void* const* d_in, const int* in_sizes, int n_in,
                              void* d_out, int out_size, void* d_ws, size_t ws_size,
                              hipStream_t stream) {
    (void)in_sizes; (void)n_in; (void)out_size; (void)ws_size;
    const float* x  = (const float*)d_in[0];
    const float* Wq = (const float*)d_in[1];
    const float* bq = (const float*)d_in[2];
    const float* Wk = (const float*)d_in[3];
    const float* bk = (const float*)d_in[4];
    const float* Wv = (const float*)d_in[5];
    const float* bv = (const float*)d_in[6];
    const float* Wo = (const float*)d_in[7];
    const float* bo = (const float*)d_in[8];

    char* ws = (char*)d_ws;
    size_t off = 0;
    __bf16* xbf  = (__bf16*)(ws + off); off += (size_t)MROWS * DM * 2;
    __bf16* Wqb  = (__bf16*)(ws + off); off += (size_t)DM * DM * 2;
    __bf16* Wkb  = (__bf16*)(ws + off); off += (size_t)DM * DM * 2;
    __bf16* Wvb  = (__bf16*)(ws + off); off += (size_t)DM * DM * 2;
    __bf16* Wob  = (__bf16*)(ws + off); off += (size_t)DM * DM * 2;
    float*  lin  = (float*)(ws + off);  off += (size_t)MROWS * DM * 4;
    __bf16* Qbf  = (__bf16*)(ws + off); off += (size_t)MROWS * DM * 2;
    __bf16* Kbf  = (__bf16*)(ws + off); off += (size_t)MROWS * DM * 2;
    __bf16* Vt   = (__bf16*)(ws + off); off += (size_t)MROWS * DM * 2;
    __bf16* Abf  = (__bf16*)(ws + off); off += (size_t)MROWS * DM * 2;

    const int nX = MROWS * DM;      // 4M elements
    const int nW = DM * DM;         // 1M elements
    cvt_bf16_kernel<<<nX / 256, 256, 0, stream>>>(x,  xbf, nX);
    cvt_bf16_kernel<<<nW / 256, 256, 0, stream>>>(Wq, Wqb, nW);
    cvt_bf16_kernel<<<nW / 256, 256, 0, stream>>>(Wk, Wkb, nW);
    cvt_bf16_kernel<<<nW / 256, 256, 0, stream>>>(Wv, Wvb, nW);
    cvt_bf16_kernel<<<nW / 256, 256, 0, stream>>>(Wo, Wob, nW);

    dim3 ggrid(MROWS / 64, DM / 128);

    // Q projection + RoPE
    gemm_bf16_kernel<<<ggrid, 256, 0, stream>>>(xbf, Wqb, bq, lin, MROWS, DM, DM);
    rope_reshape_kernel<<<(BATCH * NHEAD * SLEN * 32) / 256, 256, 0, stream>>>(lin, Qbf);
    // K projection + RoPE
    gemm_bf16_kernel<<<ggrid, 256, 0, stream>>>(xbf, Wkb, bk, lin, MROWS, DM, DM);
    rope_reshape_kernel<<<(BATCH * NHEAD * SLEN * 32) / 256, 256, 0, stream>>>(lin, Kbf);
    // V projection + transpose
    gemm_bf16_kernel<<<ggrid, 256, 0, stream>>>(xbf, Wvb, bv, lin, MROWS, DM, DM);
    v_transpose_kernel<<<nX / 256, 256, 0, stream>>>(lin, Vt);

    // Flash attention
    attn_kernel<<<dim3(BATCH * NHEAD, SLEN / 64), 128, 0, stream>>>(Qbf, Kbf, Vt, Abf);

    // Output projection -> fp32 d_out
    gemm_bf16_kernel<<<ggrid, 256, 0, stream>>>(Abf, Wob, bo, (float*)d_out, MROWS, DM, DM);
}
